// afm_57097295233531
// MI455X (gfx1250) — matrix operand order repaired and checked by reading, not compiled, not run on hardware
//
#include <hip/hip_runtime.h>
#include <hip/hip_bf16.h>
#include <math.h>

typedef __attribute__((ext_vector_type(16))) _Float16 v16h;
typedef __attribute__((ext_vector_type(8)))  _Float16 v8h;
typedef __attribute__((ext_vector_type(8)))  float    v8f;

#define Bn 512
#define Gn 48
#define Dn 256
#define HIDn 30
#define NPAIR 1128
#define NPAIR_PAD 1136
#define NTILE 71          // ceil(1128/16)
#define W1F_ELEMS (8 * 2 * 32 * 16)   // 8 K-tiles x 2 N-tiles x 32 lanes x 16 halves

__device__ __forceinline__ float sigmoidf_(float x) {
    return 1.0f / (1.0f + __expf(-x));
}

// One-block pre-pass: pack W1 (and p as column 30 of N-tile 1) into the exact
// per-lane WMMA B-fragment layout, shared by all 512 main blocks via d_ws.
// Fragment (kt,nt), lane l, elem t:  K = (t<8 ? t : t+8) + 8*(l>=16), d = kt*32+K,
// n = nt*16 + (l&15).  n<30 -> W1[d][n];  n==30 -> p[d];  n==31 -> 0.
__global__ __launch_bounds__(256) void afm_pack_kernel(
    const float* __restrict__ W1,   // [D, HID]
    const float* __restrict__ pvec, // [D]
    _Float16*    __restrict__ w1f_out)
{
    for (int idx = threadIdx.x; idx < W1F_ELEMS; idx += 256) {
        int t  = idx & 15;
        int l  = (idx >> 4) & 31;
        int nt = (idx >> 9) & 1;
        int kt = idx >> 10;
        int d  = kt * 32 + ((t < 8) ? t : t + 8) + ((l >= 16) ? 8 : 0);
        int n  = nt * 16 + (l & 15);
        float v;
        if (n < HIDn)       v = W1[d * HIDn + n];
        else if (n == HIDn) v = pvec[d];        // p rides along as B column 30
        else                v = 0.0f;
        w1f_out[idx] = (_Float16)v;
    }
}

__global__ __launch_bounds__(256) void afm_wmma_kernel(
    const int*      __restrict__ group,        // [B,G]
    const float*    __restrict__ person_embed, // [NUM_USER+1, D]
    const float*    __restrict__ first_embed,  // [NUM_USER, 1]
    const float*    __restrict__ b1,           // [HID]
    const float*    __restrict__ W2,           // [HID, 1]
    const float*    __restrict__ b2,           // [1]
    const _Float16* __restrict__ w1f_src,      // packed fragments (from pre-pass)
    float*          __restrict__ out)          // [B]
{
    __shared__ __align__(16) _Float16 sv[Gn][Dn];       // gathered embeddings (f16)
    __shared__ __align__(16) _Float16 w1f[W1F_ELEMS];   // B fragments (W1 + p column)
    __shared__ float b1s[32];
    __shared__ float w2s[32];
    __shared__ float pairA[NPAIR_PAD];
    __shared__ float pairDot[NPAIR_PAD];
    __shared__ unsigned char pi_[NPAIR_PAD];
    __shared__ unsigned char pj_[NPAIR_PAD];
    __shared__ int   sgrp[Gn];
    __shared__ float firstbuf[64];
    __shared__ float sred[256];
    __shared__ float sred2[256];

    const int tid  = threadIdx.x;
    const int b    = blockIdx.x;
    const int lane = tid & 31;
    const int wave = tid >> 5;
    const int col  = lane & 15;
    const int hi   = lane >> 4;

    // ------------- setup -------------
    if (tid < Gn) sgrp[tid] = group[b * Gn + tid];
    if (tid < 32) {
        b1s[tid] = (tid < HIDn) ? b1[tid] : 0.0f;
        w2s[tid] = (tid < HIDn) ? W2[tid] : 0.0f;   // w2s[30]=0 keeps p-column out of score
    }

    // vectorized copy of pre-packed B fragments: 16 KB as float4s
    {
        const float4* src = (const float4*)w1f_src;
        float4*       dst = (float4*)w1f;
        #pragma unroll
        for (int k = 0; k < 4; ++k) dst[tid + 256 * k] = src[tid + 256 * k];
    }

    // triu(i<j) pair index tables (row-major, matches np.triu_indices order)
    for (int pp = tid; pp < NPAIR_PAD; pp += 256) {
        int i = 0, j = 1;
        if (pp < NPAIR) {
            int rem = pp;
            while (rem >= (Gn - 1 - i)) { rem -= (Gn - 1 - i); ++i; }
            j = i + 1 + rem;
        }
        pi_[pp] = (unsigned char)i;
        pj_[pp] = (unsigned char)j;
    }
    __syncthreads();   // sgrp ready

    // gather the 48 embedding rows into LDS as f16
    for (int idx = tid; idx < Gn * Dn; idx += 256) {
        int row = idx >> 8;
        int d   = idx & 255;
        sv[row][d] = (_Float16)person_embed[(long)sgrp[row] * Dn + d];
    }
    if (tid < 64) firstbuf[tid] = (tid < Gn) ? first_embed[sgrp[tid]] : 0.0f;
    __syncthreads();

    const float bias2 = b2[0];

    // ------------- per-wave 16-pair tiles -------------
    for (int tile = wave; tile < NTILE; tile += 8) {
        const int pp = tile * 16 + col;       // this lane's pair (A-matrix row M=col)
        const int ii = pi_[pp];
        const int jj = pj_[pp];
        v8f c0 = {}; v8f c1 = {};
        #pragma unroll
        for (int kt = 0; kt < 8; ++kt) {
            // A layout: lane group hi covers K = {0..7,16..23} + 8*hi of this 32-K tile
            const int base = kt * 32 + 8 * hi;
            v8h vi0 = *(const v8h*)&sv[ii][base];
            v8h vj0 = *(const v8h*)&sv[jj][base];
            v8h vi1 = *(const v8h*)&sv[ii][base + 16];
            v8h vj1 = *(const v8h*)&sv[jj][base + 16];
            v8h e0 = vi0 * vj0;               // v_pk_mul_f16, stays in f16
            v8h e1 = vi1 * vj1;
            v16h a = __builtin_shufflevector(e0, e1,
                       0, 1, 2, 3, 4, 5, 6, 7, 8, 9, 10, 11, 12, 13, 14, 15);
            v16h bf0 = *(const v16h*)&w1f[((kt * 2 + 0) * 32 + lane) * 16];
            v16h bf1 = *(const v16h*)&w1f[((kt * 2 + 1) * 32 + lane) * 16];
            c0 = __builtin_amdgcn_wmma_f32_16x16x32_f16(false, a, false, bf0, (short)0, c0, false, false);
            c1 = __builtin_amdgcn_wmma_f32_16x16x32_f16(false, a, false, bf1, (short)0, c1, false, false);
        }

        // c0[r]: pair M=r+8*hi, n=col (0..15).  c1[r]: n=col+16 (16..31).
        // Column 30 of c1 (col==14) is e.p — the second-order dot product.
        if (col == 14) {
            #pragma unroll
            for (int r = 0; r < 8; ++r) {
                int ppw = tile * 16 + r + 8 * hi;
                if (ppw < NPAIR) pairDot[ppw] = c1[r];
            }
        }

        // attention logit: sum_n sigmoid(c+b1[n])*W2[n]; reduce over 16-lane group
        float sums[8];
        #pragma unroll
        for (int r = 0; r < 8; ++r) {
            float s0 = sigmoidf_(c0[r] + b1s[col]) * w2s[col];
            float s1 = sigmoidf_(c1[r] + b1s[col + 16]) * w2s[col + 16];
            sums[r] = s0 + s1;
        }
        #pragma unroll
        for (int msk = 1; msk < 16; msk <<= 1) {
            #pragma unroll
            for (int r = 0; r < 8; ++r) sums[r] += __shfl_xor(sums[r], msk, 32);
        }
        if (col < 8) {
            int ppw = tile * 16 + col + 8 * hi;
            if (ppw < NPAIR) pairA[ppw] = sigmoidf_(sums[col] + bias2);
        }
    }
    __syncthreads();

    // ------------- softmax over 1128 pairs + weighted sum -------------
    float m = -3.402823466e+38f;
    for (int idx = tid; idx < NPAIR; idx += 256) m = fmaxf(m, pairA[idx]);
    sred[tid] = m;
    __syncthreads();
    for (int s = 128; s > 0; s >>= 1) {
        if (tid < s) sred[tid] = fmaxf(sred[tid], sred[tid + s]);
        __syncthreads();
    }
    const float gmax = sred[0];
    __syncthreads();

    float se = 0.0f, sd = 0.0f;
    for (int idx = tid; idx < NPAIR; idx += 256) {
        float e = __expf(pairA[idx] - gmax);
        se += e;
        sd += e * pairDot[idx];
    }
    sred[tid] = se; sred2[tid] = sd;
    __syncthreads();
    for (int s = 128; s > 0; s >>= 1) {
        if (tid < s) { sred[tid] += sred[tid + s]; sred2[tid] += sred2[tid + s]; }
        __syncthreads();
    }

    if (tid == 0) {
        float second = sred2[0] / sred[0];
        float first  = 0.0f;
        #pragma unroll
        for (int g = 0; g < 64; ++g) first += firstbuf[g];
        out[b] = sigmoidf_(first + second) * 2.0f - 1.0f;
    }
}

extern "C" void kernel_launch(void* const* d_in, const int* in_sizes, int n_in,
                              void* d_out, int out_size, void* d_ws, size_t ws_size,
                              hipStream_t stream) {
    (void)in_sizes; (void)n_in; (void)ws_size; (void)out_size;
    const int*   group        = (const int*)  d_in[0];
    const float* person_embed = (const float*)d_in[1];
    const float* first_embed  = (const float*)d_in[2];
    const float* pvec         = (const float*)d_in[3];
    const float* W1           = (const float*)d_in[4];
    const float* b1           = (const float*)d_in[5];
    const float* W2           = (const float*)d_in[6];
    const float* b2           = (const float*)d_in[7];
    float* out = (float*)d_out;

    _Float16* w1f = (_Float16*)d_ws;   // 16 KB of workspace

    afm_pack_kernel<<<1, 256, 0, stream>>>(W1, pvec, w1f);
    afm_wmma_kernel<<<Bn, 256, 0, stream>>>(group, person_embed, first_embed,
                                            b1, W2, b2, w1f, out);
}
